// ContrastiveLoss_66030827208766
// MI455X (gfx1250) — matrix operand order repaired and checked by reading, not compiled, not run on hardware
//
#include <hip/hip_runtime.h>

typedef __attribute__((ext_vector_type(8)))  _Float16 v8h;
typedef __attribute__((ext_vector_type(16))) _Float16 v16h;
typedef __attribute__((ext_vector_type(8)))  float    v8f;

#define BSZ 128
#define DIM 128
#define MARGIN_F 0.2f

// ---------------------------------------------------------------------------
// f32 -> f16 convert + zero-pad rows to a multiple of 16
// in:  (BSZ, rowsIn, DIM) f32   out: (BSZ, rowsPad, DIM) f16
// ---------------------------------------------------------------------------
__global__ __launch_bounds__(256) void cvt_pad_kernel(
    const float* __restrict__ in, _Float16* __restrict__ out,
    int rowsIn, int rowsPad) {
  int idx = blockIdx.x * blockDim.x + threadIdx.x;
  int total = BSZ * rowsPad * DIM;
  if (idx >= total) return;
  int d = idx & (DIM - 1);
  int r = (idx / DIM) % rowsPad;
  int b = idx / (DIM * rowsPad);
  float v = (r < rowsIn) ? in[(b * rowsIn + r) * DIM + d] : 0.0f;
  out[idx] = (_Float16)v;
}

// ---------------------------------------------------------------------------
// t2i score kernel: one wave per (image i, caption c) pair.
//   Ab: (BSZ, OT*16, DIM) f16 padded objects
//   Bb: (BSZ, NT*16, DIM) f16 padded words
//   score(i,c) = sum_{w<WVALID} max_{o<objn[i]} <A_i[o], B_c[w]>  / capl[c]
// OT/NT = 16-row tiles in O/W.  WMMA A/B use contiguous k order (k = 16*half + s).
// ---------------------------------------------------------------------------
template <int OT, int NT, int WVALID, bool ACCUM>
__global__ __launch_bounds__(256) void t2i_kernel(
    const _Float16* __restrict__ Ab, const _Float16* __restrict__ Bb,
    const int* __restrict__ objn, const int* __restrict__ capl,
    float* __restrict__ scores) {
  const int wave = blockIdx.x * (blockDim.x >> 5) + (threadIdx.x >> 5);
  const int lane = threadIdx.x & 31;
  const int i = wave >> 7;       // image index
  const int c = wave & 127;      // caption index
  const int half = lane >> 4;    // 0: lanes 0-15, 1: lanes 16-31
  const int l16  = lane & 15;

  const _Float16* Ai = Ab + (size_t)i * (OT * 16 * DIM);
  const _Float16* Bc = Bb + (size_t)c * (NT * 16 * DIM);
  const int nvalid = objn[i];

  float colsum = 0.0f;

#pragma unroll
  for (int nt = 0; nt < NT; ++nt) {
    // B tile (32x16 f16 per K-chunk): lane n holds word nt*16+n;
    // lanes 0-15 carry K=kk*32+0..15, lanes 16-31 carry K=kk*32+16..31.
    v16h btile[4];
#pragma unroll
    for (int kk = 0; kk < 4; ++kk) {
      btile[kk] = *(const v16h*)(Bc + (nt * 16 + l16) * DIM + kk * 32 + half * 16);
    }

    float colmax = -3.0e38f;
#pragma unroll
    for (int mt = 0; mt < OT; ++mt) {
      v8f acc = {};
#pragma unroll
      for (int kk = 0; kk < 4; ++kk) {
        // A tile 16x32: lane m holds row mt*16+m.
        // lanes 0-15: VGPR0-3 <- K=kk*32+0..7,   VGPR4-7 <- K=kk*32+8..15
        // lanes16-31: VGPR0-3 <- K=kk*32+16..23, VGPR4-7 <- K=kk*32+24..31
        const _Float16* pa = Ai + (mt * 16 + l16) * DIM + kk * 32 + half * 16;
        v8h lo = *(const v8h*)(pa);
        v8h hi = *(const v8h*)(pa + 8);
        v16h a = __builtin_shufflevector(lo, hi, 0, 1, 2, 3, 4, 5, 6, 7, 8, 9,
                                         10, 11, 12, 13, 14, 15);
        acc = __builtin_amdgcn_wmma_f32_16x16x32_f16(
            false, a, false, btile[kk], (short)0, acc, false, false);
      }
      // acc VGPR r: lanes 0-15 -> (M=mt*16+r, N=nt*16+lane),
      //             lanes16-31 -> (M=mt*16+8+r, N=nt*16+lane-16)
#pragma unroll
      for (int r = 0; r < 8; ++r) {
        int m = mt * 16 + r + half * 8;
        float v = (m < nvalid) ? acc[r] : -3.0e38f;
        colmax = fmaxf(colmax, v);
      }
    }
    // combine partial column maxes held by lane n and lane n+16
    colmax = fmaxf(colmax, __shfl_xor(colmax, 16, 32));
    int col = nt * 16 + l16;
    colsum += (col < WVALID) ? colmax : 0.0f;  // skip padded words
  }

  // butterfly sum; each valid column is counted twice (lane n and n+16)
#pragma unroll
  for (int off = 16; off >= 1; off >>= 1) colsum += __shfl_xor(colsum, off, 32);

  float score = colsum * 0.5f / (float)capl[c];
  if (lane == 0) {
    if (ACCUM)
      scores[i * BSZ + c] += score;
    else
      scores[i * BSZ + c] = score;
  }
}

// ---------------------------------------------------------------------------
// Margin loss with max-violation: single block, thread t owns row t (cost_s)
// and column t (cost_im).
// ---------------------------------------------------------------------------
__global__ __launch_bounds__(128) void loss_kernel(const float* __restrict__ S,
                                                   float* __restrict__ out) {
  __shared__ float red[128];
  int t = threadIdx.x;
  float diag = S[t * BSZ + t];
  float m_row = 0.0f;  // max_j!=t relu(M + S[t][j] - diag[t])
  float m_col = 0.0f;  // max_i!=t relu(M + S[i][t] - diag[t])
  for (int j = 0; j < BSZ; ++j) {
    if (j != t) {
      m_row = fmaxf(m_row, fmaxf(MARGIN_F + S[t * BSZ + j] - diag, 0.0f));
      m_col = fmaxf(m_col, fmaxf(MARGIN_F + S[j * BSZ + t] - diag, 0.0f));
    }
  }
  red[t] = m_row + m_col;
  __syncthreads();
  for (int s = 64; s >= 1; s >>= 1) {
    if (t < s) red[t] += red[t + s];
    __syncthreads();
  }
  if (t == 0) out[0] = red[0];
}

// ---------------------------------------------------------------------------
extern "C" void kernel_launch(void* const* d_in, const int* in_sizes, int n_in,
                              void* d_out, int out_size, void* d_ws,
                              size_t ws_size, hipStream_t stream) {
  (void)in_sizes; (void)n_in; (void)out_size; (void)ws_size;

  const float* im   = (const float*)d_in[0];  // (128,36,128)
  const int*   im_l = (const int*)d_in[1];
  const float* s    = (const float*)d_in[2];  // (128,50,128)
  const int*   s_l  = (const int*)d_in[3];
  const float* pred   = (const float*)d_in[4];  // (128,25,128)
  const int*   pred_l = (const int*)d_in[5];
  // d_in[6], d_in[7] (cap_o_pred / cap_o_l) unused by reference
  const float* crp   = (const float*)d_in[8];  // (128,30,128)
  const int*   crp_l = (const int*)d_in[9];

  // workspace carve (256B-aligned segments)
  char* w = (char*)d_ws;
  float*     scores = (float*)w;      w += BSZ * BSZ * sizeof(float);      // 64 KB
  _Float16*  imb    = (_Float16*)w;   w += (size_t)BSZ * 48 * DIM * 2;     // 1.5 MB
  _Float16*  sb     = (_Float16*)w;   w += (size_t)BSZ * 64 * DIM * 2;     // 2.0 MB
  _Float16*  pb     = (_Float16*)w;   w += (size_t)BSZ * 32 * DIM * 2;     // 1.0 MB
  _Float16*  cb     = (_Float16*)w;                                        // 1.0 MB

  // f32 -> f16 padded copies (memory-bound, ~15 MB total traffic)
  cvt_pad_kernel<<<(BSZ * 48 * DIM + 255) / 256, 256, 0, stream>>>(im, imb, 36, 48);
  cvt_pad_kernel<<<(BSZ * 64 * DIM + 255) / 256, 256, 0, stream>>>(s, sb, 50, 64);
  cvt_pad_kernel<<<(BSZ * 32 * DIM + 255) / 256, 256, 0, stream>>>(pred, pb, 25, 32);
  cvt_pad_kernel<<<(BSZ * 32 * DIM + 255) / 256, 256, 0, stream>>>(crp, cb, 30, 32);

  // 16384 (i,c) pairs, 8 waves (256 threads) per block -> 2048 blocks
  t2i_kernel<3, 4, 50, false><<<2048, 256, 0, stream>>>(imb, sb, im_l, s_l, scores);
  t2i_kernel<2, 2, 30, true ><<<2048, 256, 0, stream>>>(pb, cb, pred_l, crp_l, scores);

  loss_kernel<<<1, 128, 0, stream>>>(scores, (float*)d_out);
}
